// TransformerBlock_86019605004850
// MI455X (gfx1250) — matrix operand order repaired and checked by reading, not compiled, not run on hardware
//
#include <hip/hip_runtime.h>
#include <hip/hip_bf16.h>

typedef __bf16 bf16_t;
typedef __attribute__((ext_vector_type(16))) __bf16 v16bf;
typedef __attribute__((ext_vector_type(4)))  __bf16 v4bf;
typedef __attribute__((ext_vector_type(8)))  float  v8f;

#define D_MODEL 1024
#define N_HEADS 16
#define HEAD_DIM 64
#define D_FF 4096
#define BATCH 2
#define SEQ 2048
#define LN_EPS 1e-5f

// Load a 16-element bf16 fragment as two 16B chunks: p[0..7] and p[off2..off2+7].
static __device__ __forceinline__ v16bf load2x16(const bf16_t* p, int off2) {
    v16bf r;
    ((float4*)&r)[0] = *(const float4*)(p);
    ((float4*)&r)[1] = *(const float4*)(p + off2);
    return r;
}

// ---------------------------------------------------------------------------
// f32 -> bf16 elementwise (4 elems/thread, vectorized).
// ---------------------------------------------------------------------------
__global__ __launch_bounds__(256) void cvt_bf16_kernel(
    const float* __restrict__ in, bf16_t* __restrict__ out)
{
    size_t base = ((size_t)blockIdx.x * 256 + threadIdx.x) * 4;
    float4 f = *(const float4*)(in + base);
    v4bf h = { (bf16_t)f.x, (bf16_t)f.y, (bf16_t)f.z, (bf16_t)f.w };
    *(v4bf*)(out + base) = h;
}

// ---------------------------------------------------------------------------
// Wt[n][k] = (bf16) W[k][n]  — 32x32 LDS tile transpose, block 256 = 32x8.
// ---------------------------------------------------------------------------
__global__ __launch_bounds__(256) void transpose_bf16_kernel(
    const float* __restrict__ W, bf16_t* __restrict__ Wt, int K, int N)
{
    __shared__ float tile[32][33];
    const int nb = blockIdx.x * 32;
    const int kb = blockIdx.y * 32;
    const int tx = threadIdx.x & 31;
    const int ty = threadIdx.x >> 5;   // 0..7
#pragma unroll
    for (int i = 0; i < 32; i += 8)
        tile[ty + i][tx] = W[(size_t)(kb + ty + i) * N + nb + tx];
    __syncthreads();
#pragma unroll
    for (int i = 0; i < 32; i += 8)
        Wt[(size_t)(nb + ty + i) * K + kb + tx] = (bf16_t)tile[tx][ty + i];
}

// ---------------------------------------------------------------------------
// LDS-free GEMM: C[M,N] = A[M,K](bf16) @ Bt[N,K](bf16)^T + bias.
// 256 threads = 8 waves (2x4); each wave computes 64x32 = 4x2 WMMA tiles.
// Fragments load straight from global (2 x b128 each); operand reuse via
// WGP$/L2 (whole B fits in the 192MB L2). No barriers in the hot loop.
// ---------------------------------------------------------------------------
template<bool RELU, bool OUTBF>
__global__ __launch_bounds__(256) void gemm_tn_kernel(
    const bf16_t* __restrict__ A,    // [M][K]
    const bf16_t* __restrict__ Bt,   // [N][K]
    const float*  __restrict__ bias,
    float* __restrict__ Cf, bf16_t* __restrict__ Cb,
    int M, int N, int K)
{
    const int lane  = threadIdx.x & 31;
    const int wave  = threadIdx.x >> 5;
    const int waveM = wave >> 2;                // 0..1
    const int waveN = wave & 3;                 // 0..3
    const int bM    = blockIdx.y * 128;
    const int bN    = blockIdx.x * 128;

    const int n15   = lane & 15;
    const int kb0   = (lane < 16) ? 0 : 8;      // 16-bit A-fragment K base
    const int kbB   = (lane < 16) ? 0 : 8;      // 16-bit B-fragment K base
    const int rbase = (lane >= 16) ? 8 : 0;     // C-layout row offset

    const bf16_t* a0 = A  + (size_t)(bM + waveM * 64 + n15) * K;
    const bf16_t* b0 = Bt + (size_t)(bN + waveN * 32 + n15) * K;

    const v8f vz = {0.f,0.f,0.f,0.f,0.f,0.f,0.f,0.f};
    v8f acc[4][2];
#pragma unroll
    for (int i = 0; i < 4; ++i)
#pragma unroll
        for (int j = 0; j < 2; ++j) acc[i][j] = vz;

    for (int k0 = 0; k0 < K; k0 += 32) {
        v16bf bfr[2];
#pragma unroll
        for (int tn = 0; tn < 2; ++tn)
            bfr[tn] = load2x16(b0 + (size_t)tn * 16 * K + k0 + kbB, 16);
#pragma unroll
        for (int tm = 0; tm < 4; ++tm) {
            v16bf af = load2x16(a0 + (size_t)tm * 16 * K + k0 + kb0, 16);
#pragma unroll
            for (int tn = 0; tn < 2; ++tn)
                acc[tm][tn] = __builtin_amdgcn_wmma_f32_16x16x32_bf16(
                    false, af, false, bfr[tn], (short)0, acc[tm][tn], false, false);
        }
    }

#pragma unroll
    for (int tm = 0; tm < 4; ++tm) {
#pragma unroll
        for (int tn = 0; tn < 2; ++tn) {
            int col = bN + waveN * 32 + tn * 16 + n15;
            float bval = bias[col];
#pragma unroll
            for (int r = 0; r < 8; ++r) {
                int row = bM + waveM * 64 + tm * 16 + rbase + r;
                float v = acc[tm][tn][r] + bval;
                if (RELU) v = fmaxf(v, 0.f);
                if (OUTBF) Cb[(size_t)row * N + col] = (bf16_t)v;
                else       Cf[(size_t)row * N + col] = v;
            }
        }
    }
}

// ---------------------------------------------------------------------------
// PoPE: mu = softplus(q/k); phases = pos*freq + bias; emit bf16
//   q_pope/k_pope: [bh][s][128]  (cos part d, sin part d+64)
//   v_t:           [bh][d][s]    (transposed for P@V B-fragments)
// ---------------------------------------------------------------------------
__global__ __launch_bounds__(256) void pope_kernel(
    const float* __restrict__ q, const float* __restrict__ k,
    const float* __restrict__ v, const int* __restrict__ positions,
    const float* __restrict__ phase_bias, const float* __restrict__ freqs,
    bf16_t* __restrict__ qp, bf16_t* __restrict__ kp, bf16_t* __restrict__ vt)
{
    int t    = blockIdx.x * 256 + threadIdx.x;
    int d    = t & (HEAD_DIM - 1);
    int rest = t >> 6;
    int s    = rest & (SEQ - 1);
    int bh   = rest >> 11;
    int b    = bh >> 4, h = bh & 15;

    size_t src = ((size_t)(b * SEQ + s)) * D_MODEL + h * HEAD_DIM + d;
    float qv = q[src], kv = k[src], vv = v[src];
    float muq = (qv > 20.f) ? qv : log1pf(expf(qv));
    float muk = (kv > 20.f) ? kv : log1pf(expf(kv));

    float ph = (float)positions[s] * freqs[d] + phase_bias[h * HEAD_DIM + d];
    float sp, cp;
    sincosf(ph, &sp, &cp);

    size_t dst = ((size_t)bh * SEQ + s) * 128 + d;
    qp[dst]      = (bf16_t)(muq * cp);
    qp[dst + 64] = (bf16_t)(muq * sp);
    kp[dst]      = (bf16_t)(muk * cp);
    kp[dst + 64] = (bf16_t)(muk * sp);
    vt[((size_t)bh * HEAD_DIM + d) * SEQ + s] = (bf16_t)vv;
}

// ---------------------------------------------------------------------------
// Flash-style causal attention with softmax1: out = P@V / (1 + sum(exp)).
// Block = 8 waves, each wave owns 16 query rows; kv tiles of 64 keys.
// Output written as bf16 [B*S][D_MODEL] (A-operand of the Wo GEMM).
// ---------------------------------------------------------------------------
__global__ __launch_bounds__(256) void attn_kernel(
    const bf16_t* __restrict__ qp, const bf16_t* __restrict__ kp,
    const bf16_t* __restrict__ vt, bf16_t* __restrict__ out)
{
    __shared__ bf16_t wlds[8][16 * 64];    // per-wave weight slice (C->A relayout)

    const int tid   = threadIdx.x;
    const int lane  = tid & 31;
    const int wave  = tid >> 5;
    const int bh    = blockIdx.y;
    const int b     = bh >> 4, h = bh & 15;
    const int qb    = blockIdx.x * 128 + wave * 16;
    const int n15   = lane & 15;
    const int kb0   = (lane < 16) ? 0 : 8;
    const int kbB   = (lane < 16) ? 0 : 8;
    const int rbase = (lane >= 16) ? 8 : 0;

    const bf16_t* qbase = qp + (size_t)bh * SEQ * 128;
    const bf16_t* kbase = kp + (size_t)bh * SEQ * 128;
    const bf16_t* vbase = vt + (size_t)bh * HEAD_DIM * SEQ;
    bf16_t* myw = &wlds[wave][0];

    v16bf qfrag[4];
#pragma unroll
    for (int kk = 0; kk < 4; ++kk) {
        const bf16_t* p = qbase + (size_t)(qb + n15) * 128 + kk * 32 + kb0;
        qfrag[kk] = load2x16(p, 16);
    }

    const v8f vz = {0.f,0.f,0.f,0.f,0.f,0.f,0.f,0.f};
    v8f acc[4];
#pragma unroll
    for (int tn = 0; tn < 4; ++tn) acc[tn] = vz;
    float mrow[8], lrow[8];
#pragma unroll
    for (int r = 0; r < 8; ++r) { mrow[r] = -__builtin_inff(); lrow[r] = 0.f; }

    const float scale  = 0.08838834764831845f;     // 1/sqrt(2*HEAD_DIM)
    const int   ntiles = (qb + 15) / 64 + 1;       // causal: kv0 <= qb always

    for (int t = 0; t < ntiles; ++t) {
        const int kv0 = t * 64;

        // Scores S = Q @ K^T for 16x64 tile (4 N-tiles x 4 K-steps).
        v8f sc[4];
#pragma unroll
        for (int tn = 0; tn < 4; ++tn) {
            v8f s = vz;
#pragma unroll
            for (int kk = 0; kk < 4; ++kk) {
                const bf16_t* p = kbase + (size_t)(kv0 + tn * 16 + n15) * 128 + kk * 32 + kbB;
                v16bf kf = load2x16(p, 16);
                s = __builtin_amdgcn_wmma_f32_16x16x32_bf16(
                    false, qfrag[kk], false, kf, (short)0, s, false, false);
            }
            sc[tn] = s;
        }

        // Scale + causal mask + online softmax1 update (per-row, half-wave reduce).
#pragma unroll
        for (int r = 0; r < 8; ++r) {
            const int qrow = qb + rbase + r;
            float mx = -__builtin_inff();
#pragma unroll
            for (int tn = 0; tn < 4; ++tn) {
                int key = kv0 + tn * 16 + n15;
                float sv = sc[tn][r] * scale;
                if (key > qrow) sv = -__builtin_inff();
                sc[tn][r] = sv;
                mx = fmaxf(mx, sv);
            }
#pragma unroll
            for (int off = 1; off < 16; off <<= 1)
                mx = fmaxf(mx, __shfl_xor(mx, off, 32));

            float mnew = fmaxf(mrow[r], mx);
            float corr = __expf(fminf(mrow[r] - mnew, 0.f));
            mrow[r] = mnew;
            lrow[r] *= corr;
#pragma unroll
            for (int tn = 0; tn < 4; ++tn) acc[tn][r] *= corr;

            float lsum = 0.f;
#pragma unroll
            for (int tn = 0; tn < 4; ++tn) {
                float w = __expf(sc[tn][r] - mnew);
                myw[(rbase + r) * 64 + tn * 16 + n15] = (bf16_t)w;
                lsum += w;
            }
#pragma unroll
            for (int off = 1; off < 16; off <<= 1)
                lsum += __shfl_xor(lsum, off, 32);
            lrow[r] += lsum;
        }

        // acc += W(16x64) @ V(64 keys x 64 dims); W re-read from LDS in A layout.
        v16bf wf[2];
#pragma unroll
        for (int kk = 0; kk < 2; ++kk) {
            const bf16_t* p = myw + n15 * 64 + kk * 32 + kb0;
            wf[kk] = load2x16(p, 16);
        }
#pragma unroll
        for (int tn = 0; tn < 4; ++tn) {
#pragma unroll
            for (int kk = 0; kk < 2; ++kk) {
                const bf16_t* p = vbase + (size_t)(tn * 16 + n15) * SEQ + kv0 + kk * 32 + kbB;
                v16bf vfr = load2x16(p, 16);
                acc[tn] = __builtin_amdgcn_wmma_f32_16x16x32_bf16(
                    false, wf[kk], false, vfr, (short)0, acc[tn], false, false);
            }
        }
    }

    // softmax1 epilogue: divide by (1 + l), scatter back as bf16 (B,S,D).
#pragma unroll
    for (int tn = 0; tn < 4; ++tn) {
#pragma unroll
        for (int r = 0; r < 8; ++r) {
            int row = qb + rbase + r;
            float o = acc[tn][r] / (1.f + lrow[r]);
            out[((size_t)b * SEQ + row) * D_MODEL + h * HEAD_DIM + tn * 16 + n15] = (bf16_t)o;
        }
    }
}

// ---------------------------------------------------------------------------
// out = LayerNorm(a + b) * g + beta; optional bf16 copy of out (FFN input).
// ---------------------------------------------------------------------------
template<bool WB>
__global__ __launch_bounds__(256) void add_ln_kernel(
    const float* __restrict__ a, const float* __restrict__ b,
    const float* __restrict__ g, const float* __restrict__ beta,
    float* __restrict__ out, bf16_t* __restrict__ outb)
{
    __shared__ float red[256];
    const int row = blockIdx.x;
    const int tid = threadIdx.x;
    const float* pa = a + (size_t)row * D_MODEL;
    const float* pb = b + (size_t)row * D_MODEL;

    float x[4];
    float s = 0.f;
#pragma unroll
    for (int i = 0; i < 4; ++i) {
        x[i] = pa[tid + i * 256] + pb[tid + i * 256];
        s += x[i];
    }
    red[tid] = s; __syncthreads();
    for (int off = 128; off > 0; off >>= 1) {
        if (tid < off) red[tid] += red[tid + off];
        __syncthreads();
    }
    float mu = red[0] * (1.f / D_MODEL);
    __syncthreads();

    float vs = 0.f;
#pragma unroll
    for (int i = 0; i < 4; ++i) { float d = x[i] - mu; vs += d * d; }
    red[tid] = vs; __syncthreads();
    for (int off = 128; off > 0; off >>= 1) {
        if (tid < off) red[tid] += red[tid + off];
        __syncthreads();
    }
    float inv = rsqrtf(red[0] * (1.f / D_MODEL) + LN_EPS);

#pragma unroll
    for (int i = 0; i < 4; ++i) {
        int c = tid + i * 256;
        float o = (x[i] - mu) * inv * g[c] + beta[c];
        out[(size_t)row * D_MODEL + c] = o;
        if (WB) outb[(size_t)row * D_MODEL + c] = (bf16_t)o;
    }
}

// ---------------------------------------------------------------------------
extern "C" void kernel_launch(void* const* d_in, const int* in_sizes, int n_in,
                              void* d_out, int out_size, void* d_ws, size_t ws_size,
                              hipStream_t stream)
{
    const float* x          = (const float*)d_in[0];
    const int*   positions  = (const int*)d_in[1];
    const float* Wq         = (const float*)d_in[2];
    const float* bq         = (const float*)d_in[3];
    const float* Wk         = (const float*)d_in[4];
    const float* bk         = (const float*)d_in[5];
    const float* Wv         = (const float*)d_in[6];
    const float* bv         = (const float*)d_in[7];
    const float* Wo         = (const float*)d_in[8];
    const float* bo         = (const float*)d_in[9];
    const float* phase_bias = (const float*)d_in[10];
    const float* freqs      = (const float*)d_in[11];
    const float* W1         = (const float*)d_in[12];
    const float* b1         = (const float*)d_in[13];
    const float* W2         = (const float*)d_in[14];
    const float* b2         = (const float*)d_in[15];
    const float* g1         = (const float*)d_in[16];
    const float* beta1      = (const float*)d_in[17];
    const float* g2         = (const float*)d_in[18];
    const float* beta2      = (const float*)d_in[19];
    float* out = (float*)d_out;

    char* ws = (char*)d_ws;
    const size_t MB = 1024ull * 1024ull;
    float*  qf     = (float*) (ws +   0 * MB);  // 16 MB -> reused as proj
    float*  kf     = (float*) (ws +  16 * MB);  // 16 MB -> reused as ffy
    float*  vf     = (float*) (ws +  32 * MB);  // 16 MB -> reused as x1
    bf16_t* qpb    = (bf16_t*)(ws +  48 * MB);  // 16 MB
    bf16_t* kpb    = (bf16_t*)(ws +  64 * MB);  // 16 MB
    bf16_t* vtb    = (bf16_t*)(ws +  80 * MB);  //  8 MB
    bf16_t* xb     = (bf16_t*)(ws +  88 * MB);  //  8 MB
    bf16_t* wqt    = (bf16_t*)(ws +  96 * MB);  //  2 MB
    bf16_t* wkt    = (bf16_t*)(ws +  98 * MB);  //  2 MB
    bf16_t* wvt    = (bf16_t*)(ws + 100 * MB);  //  2 MB
    bf16_t* wot    = (bf16_t*)(ws + 102 * MB);  //  2 MB
    bf16_t* w1t    = (bf16_t*)(ws + 104 * MB);  //  8 MB
    bf16_t* w2t    = (bf16_t*)(ws + 112 * MB);  //  8 MB
    bf16_t* attnb  = (bf16_t*)(ws + 120 * MB);  //  8 MB
    bf16_t* x1b    = (bf16_t*)(ws + 128 * MB);  //  8 MB
    bf16_t* ffhb   = (bf16_t*)(ws + 136 * MB);  // 32 MB   (total 168 MB)
    float*  proj = qf;   // reuse after PoPE consumed q
    float*  ffy  = kf;   // reuse after PoPE consumed k
    float*  x1   = vf;   // reuse after PoPE consumed v

    const int M = BATCH * SEQ;                  // 4096
    dim3 blk(256);
    dim3 gD(D_MODEL / 128, M / 128);            // 8 x 32
    dim3 gF(D_FF / 128, M / 128);               // 32 x 32

    // --- one-time operand prep: activations + transposed weights in bf16 ---
    cvt_bf16_kernel<<<dim3((M * D_MODEL) / 1024), blk, 0, stream>>>(x, xb);
    transpose_bf16_kernel<<<dim3(D_MODEL/32, D_MODEL/32), blk, 0, stream>>>(Wq, wqt, D_MODEL, D_MODEL);
    transpose_bf16_kernel<<<dim3(D_MODEL/32, D_MODEL/32), blk, 0, stream>>>(Wk, wkt, D_MODEL, D_MODEL);
    transpose_bf16_kernel<<<dim3(D_MODEL/32, D_MODEL/32), blk, 0, stream>>>(Wv, wvt, D_MODEL, D_MODEL);
    transpose_bf16_kernel<<<dim3(D_MODEL/32, D_MODEL/32), blk, 0, stream>>>(Wo, wot, D_MODEL, D_MODEL);
    transpose_bf16_kernel<<<dim3(D_FF/32,    D_MODEL/32), blk, 0, stream>>>(W1, w1t, D_MODEL, D_FF);
    transpose_bf16_kernel<<<dim3(D_MODEL/32, D_FF/32),    blk, 0, stream>>>(W2, w2t, D_FF, D_MODEL);

    // --- QKV projections (f32 out for PoPE) ---
    gemm_tn_kernel<false,false><<<gD, blk, 0, stream>>>(xb, wqt, bq, qf, nullptr, M, D_MODEL, D_MODEL);
    gemm_tn_kernel<false,false><<<gD, blk, 0, stream>>>(xb, wkt, bk, kf, nullptr, M, D_MODEL, D_MODEL);
    gemm_tn_kernel<false,false><<<gD, blk, 0, stream>>>(xb, wvt, bv, vf, nullptr, M, D_MODEL, D_MODEL);

    // --- PoPE + layout transform ---
    pope_kernel<<<dim3((BATCH * N_HEADS * SEQ * HEAD_DIM) / 256), blk, 0, stream>>>(
        qf, kf, vf, positions, phase_bias, freqs, qpb, kpb, vtb);

    // --- causal attention with softmax1 (bf16 out) ---
    attn_kernel<<<dim3(SEQ / 128, BATCH * N_HEADS), blk, 0, stream>>>(qpb, kpb, vtb, attnb);

    // --- output projection + residual LN (x1 f32 + bf16 copy) ---
    gemm_tn_kernel<false,false><<<gD, blk, 0, stream>>>(attnb, wot, bo, proj, nullptr, M, D_MODEL, D_MODEL);
    add_ln_kernel<true><<<dim3(M), blk, 0, stream>>>(x, proj, g1, beta1, x1, x1b);

    // --- FFN ---
    gemm_tn_kernel<true, true ><<<gF, blk, 0, stream>>>(x1b, w1t, b1, nullptr, ffhb, M, D_FF, D_MODEL);
    gemm_tn_kernel<false,false><<<gD, blk, 0, stream>>>(ffhb, w2t, b2, ffy, nullptr, M, D_MODEL, D_FF);
    add_ln_kernel<false><<<dim3(M), blk, 0, stream>>>(x1, ffy, g2, beta2, out, nullptr);
}
